// MultiHeadSelfAttention_73005854097681
// MI455X (gfx1250) — matrix operand order repaired and checked by reading, not compiled, not run on hardware
//
#include <hip/hip_runtime.h>
#include <hip/hip_bf16.h>

// ---------------------------------------------------------------------------
// MHA forward for MI455X (gfx1250, wave32, WMMA + Tensor Data Mover).
// Pipeline: f32->bf16 | QKV gemm (wmma bf16, TDM double-buffered staging) |
// RoPE | flash-attn (wmma bf16, online softmax, TDM K tiles) | Wo gemm -> f32.
// ---------------------------------------------------------------------------

#define BSZ    2
#define SEQ    2048
#define DMODEL 1024
#define NHEAD  16
#define DK     64
#define MROWS  (BSZ * SEQ)  // 4096

typedef __bf16 v16bf __attribute__((ext_vector_type(16)));
typedef __bf16 v8bf  __attribute__((ext_vector_type(8)));
typedef float  v8f   __attribute__((ext_vector_type(8)));
typedef unsigned int v4u __attribute__((ext_vector_type(4)));
typedef int v4i __attribute__((ext_vector_type(4)));
typedef int v8i __attribute__((ext_vector_type(8)));

__device__ __forceinline__ v8f vzero8() {
  v8f z;
#pragma unroll
  for (int i = 0; i < 8; ++i) z[i] = 0.0f;
  return z;
}

__device__ __forceinline__ v16bf ld_frag16(const __bf16* lo_p, const __bf16* hi_p) {
  v8bf lo = *reinterpret_cast<const v8bf*>(lo_p);
  v8bf hi = *reinterpret_cast<const v8bf*>(hi_p);
  return __builtin_shufflevector(lo, hi, 0, 1, 2, 3, 4, 5, 6, 7,
                                 8, 9, 10, 11, 12, 13, 14, 15);
}

__device__ __forceinline__ v8f wmma_bf16(v16bf a, v16bf b, v8f c) {
  return __builtin_amdgcn_wmma_f32_16x16x32_bf16(
      false, a, false, b, (short)0, c, false, false);
}

// ---------------------------------------------------------------------------
// Tensor Data Mover: 2-D bf16 tile, global -> LDS, optional LDS row padding.
// D# packing per CDNA5 ISA.
// pad_interval_code: row length = 2^(code+1) DWORDs; pad_amount_code: pad =
// (code+1) DWORDs appended per row in LDS.
// ---------------------------------------------------------------------------
__device__ __forceinline__ void tdm_load_2d_bf16(
    const __bf16* gsrc, unsigned lds_off,
    unsigned tile_d0, unsigned tile_d1,
    unsigned tensor_d0, unsigned tensor_d1,
    unsigned long long row_stride_elems,
    unsigned pad_en, unsigned pad_interval_code, unsigned pad_amount_code) {
  const unsigned long long ga = (unsigned long long)(uintptr_t)gsrc;
  v4u g0;
  g0[0] = 1u;                                                // count=1, user
  g0[1] = lds_off;                                           // LDS byte addr
  g0[2] = (unsigned)ga;                                      // global lo
  g0[3] = (unsigned)((ga >> 32) & 0x01FFFFFFu) | (2u << 30); // hi + type=2
  v8i g1;
  g1[0] = (int)((1u << 16) |                       // data_size = 2 bytes
                (pad_en << 20) | (pad_interval_code << 22) |
                (pad_amount_code << 25));
  g1[1] = (int)((tensor_d0 & 0xFFFFu) << 16);      // tensor_dim0 [79:48]
  g1[2] = (int)(((tensor_d0 >> 16) & 0xFFFFu) |
                ((tensor_d1 & 0xFFFFu) << 16));    // tensor_dim1 [111:80]
  g1[3] = (int)(((tensor_d1 >> 16) & 0xFFFFu) |
                ((tile_d0 & 0xFFFFu) << 16));      // tile_dim0 [127:112]
  g1[4] = (int)(tile_d1 & 0xFFFFu);                // tile_dim1; tile_dim2=0
  g1[5] = (int)(unsigned)(row_stride_elems);       // dim0_stride lo32
  g1[6] = (int)(unsigned)((row_stride_elems >> 32) & 0xFFFFu);
  g1[7] = 0;
  v4i gz = (v4i){0, 0, 0, 0};
#if __has_include(<hip/amd_detail/amd_gfx1250_TDM.h>)
  v8i gz8 = (v8i){0, 0, 0, 0, 0, 0, 0, 0};
  __builtin_amdgcn_tensor_load_to_lds(g0, g1, gz, gz, gz8, 0);
#else
  __builtin_amdgcn_tensor_load_to_lds(g0, g1, gz, gz, 0);
#endif
}

// ---------------------------------------------------------------------------
// fp32 -> bf16 conversion
// ---------------------------------------------------------------------------
__global__ void f32_to_bf16_kernel(const float* __restrict__ in,
                                   __bf16* __restrict__ out, int n) {
  int i = blockIdx.x * blockDim.x + threadIdx.x;
  if (i < n) out[i] = (__bf16)in[i];
}

// ---------------------------------------------------------------------------
// NT GEMM: C[M,N] = A[M,K] * W[N,K]^T, bf16 in, wmma f32 accumulate.
// 128x128 block tile, BK=32, TDM double-buffered LDS staging.
// ---------------------------------------------------------------------------
template <bool OUTF32>
__global__ __launch_bounds__(256) void gemm_nt_kernel(
    const __bf16* __restrict__ A, const __bf16* __restrict__ W,
    void* __restrict__ Cv, int Mn, int Nn, int Kn) {
  constexpr int LDT = 40;  // 32 data + 8 pad halfs per row (TDM pad 3/3)
  __shared__ __bf16 As[2][128 * LDT];
  __shared__ __bf16 Bs[2][128 * LDT];

  const int tid = threadIdx.x;
  const int wave = tid >> 5;
  const int lane = tid & 31;
  const int half = lane >> 4;
  const int l16 = lane & 15;
  const int waveM = wave & 1;
  const int waveN = wave >> 1;
  const int mBase = blockIdx.y * 128;
  const int nBase = blockIdx.x * 128;
  const int nIter = Kn >> 5;

  unsigned as_off[2], bs_off[2];
#pragma unroll
  for (int s = 0; s < 2; ++s) {
    as_off[s] = (unsigned)(uintptr_t)(&As[s][0]);
    bs_off[s] = (unsigned)(uintptr_t)(&Bs[s][0]);
  }

  v8f acc[4][2];
#pragma unroll
  for (int i = 0; i < 4; ++i)
#pragma unroll
    for (int j = 0; j < 2; ++j) acc[i][j] = vzero8();

  // Prologue: DMA tile 0 into buffer 0.
  if (wave == 0) {
    tdm_load_2d_bf16(A + (size_t)mBase * Kn, as_off[0], 32, 128,
                     (unsigned)Kn, (unsigned)Mn, (unsigned)Kn, 1, 3, 3);
    tdm_load_2d_bf16(W + (size_t)nBase * Kn, bs_off[0], 32, 128,
                     (unsigned)Kn, (unsigned)Nn, (unsigned)Kn, 1, 3, 3);
  }

  for (int it = 0; it < nIter; ++it) {
    const int cur = it & 1;
    if (wave == 0) {
      if (it + 1 < nIter) {  // overlap: DMA next tile, then wait current done
        const int kk = (it + 1) << 5;
        tdm_load_2d_bf16(A + (size_t)mBase * Kn + kk, as_off[cur ^ 1], 32, 128,
                         (unsigned)Kn, (unsigned)Mn, (unsigned)Kn, 1, 3, 3);
        tdm_load_2d_bf16(W + (size_t)nBase * Kn + kk, bs_off[cur ^ 1], 32, 128,
                         (unsigned)Kn, (unsigned)Nn, (unsigned)Kn, 1, 3, 3);
        __builtin_amdgcn_s_wait_tensorcnt(2);  // tile `it` (first 2 ops) done
      } else {
        __builtin_amdgcn_s_wait_tensorcnt(0);
      }
    }
    __syncthreads();  // tile `it` visible to all waves

    const __bf16* Ab = &As[cur][0];
    const __bf16* Bb = &Bs[cur][0];
    v16bf af[4];
#pragma unroll
    for (int i = 0; i < 4; ++i) {
      const __bf16* pr = Ab + (waveM * 64 + i * 16 + l16) * LDT;
      af[i] = ld_frag16(pr + half * 8, pr + 16 + half * 8);
    }
    v16bf bfr[2];
#pragma unroll
    for (int j = 0; j < 2; ++j) {
      const __bf16* pr = Bb + (waveN * 32 + j * 16 + l16) * LDT + half * 8;
      bfr[j] = ld_frag16(pr, pr + 16);
    }
#pragma unroll
    for (int i = 0; i < 4; ++i)
#pragma unroll
      for (int j = 0; j < 2; ++j) acc[i][j] = wmma_bf16(af[i], bfr[j], acc[i][j]);

    __syncthreads();  // all reads of this buffer done before TDM reuses it
  }

  float* Cf = reinterpret_cast<float*>(Cv);
  __bf16* Cb = reinterpret_cast<__bf16*>(Cv);
#pragma unroll
  for (int i = 0; i < 4; ++i)
#pragma unroll
    for (int j = 0; j < 2; ++j) {
      const int m0 = mBase + waveM * 64 + i * 16 + half * 8;
      const int n = nBase + waveN * 32 + j * 16 + l16;
#pragma unroll
      for (int r = 0; r < 8; ++r) {
        const float v = acc[i][j][r];
        if (OUTF32)
          Cf[(size_t)(m0 + r) * Nn + n] = v;
        else
          Cb[(size_t)(m0 + r) * Nn + n] = (__bf16)v;
      }
    }
}

// ---------------------------------------------------------------------------
// RoPE: src [B,S,H*DK] bf16 -> dst [B,H,S,DK] bf16, interleaved-pair rotation.
// ---------------------------------------------------------------------------
__global__ void rope_kernel(const __bf16* __restrict__ src,
                            __bf16* __restrict__ dst,
                            const int* __restrict__ tpos) {
  const int idx = blockIdx.x * blockDim.x + threadIdx.x;
  const int total = BSZ * NHEAD * SEQ * (DK / 2);
  if (idx >= total) return;
  const int p = idx & 31;
  int rest = idx >> 5;
  const int h = rest & (NHEAD - 1);
  rest >>= 4;
  const int s = rest & (SEQ - 1);
  const int b = rest >> 11;

  const float inv_freq = __powf(10000.0f, -(float)(2 * p) / (float)DK);
  const float ang = (float)tpos[s] * inv_freq;
  float sn, cs;
  __sincosf(ang, &sn, &cs);

  const __bf16* sp = src + ((size_t)(b * SEQ + s)) * DMODEL + h * DK + 2 * p;
  const float x0 = (float)sp[0];
  const float x1 = (float)sp[1];
  __bf16* dp = dst + (((size_t)(b * NHEAD + h) * SEQ) + s) * DK + 2 * p;
  dp[0] = (__bf16)(x0 * cs - x1 * sn);
  dp[1] = (__bf16)(x1 * cs + x0 * sn);
}

// ---------------------------------------------------------------------------
// Flash attention (causal). Q,K: [B,H,S,DK] bf16, V: [B,S,H*DK] bf16.
// 128 q-rows/block, 8 waves x 16 rows; kv tiles of 64. K tile via TDM (padded
// stride 72 halfs), V staged transposed manually. Online softmax in f32.
// ---------------------------------------------------------------------------
__global__ __launch_bounds__(256) void flash_attn_kernel(
    const __bf16* __restrict__ Q, const __bf16* __restrict__ K,
    const __bf16* __restrict__ V, __bf16* __restrict__ O) {
  constexpr int LDT = 72;  // 64 data + 8 pad halfs (TDM pad 4/3)
  __shared__ __bf16 Ks[64 * LDT];      // [kv][d]
  __shared__ __bf16 Vt[64 * LDT];      // [d][kv] transposed
  __shared__ __bf16 Pl[8 * 16 * LDT];  // per-wave P tiles

  const int tid = threadIdx.x;
  const int wave = tid >> 5;
  const int lane = tid & 31;
  const int half = lane >> 4;
  const int l16 = lane & 15;
  const int bh = blockIdx.y;
  const int b = bh >> 4;
  const int h = bh & (NHEAD - 1);
  const int qbase = blockIdx.x * 128;

  const __bf16* Qb = Q + (size_t)bh * SEQ * DK;
  const __bf16* Kb = K + (size_t)bh * SEQ * DK;
  const __bf16* Vb = V + (size_t)b * SEQ * DMODEL + h * DK;
  const unsigned ks_off = (unsigned)(uintptr_t)(&Ks[0]);

  const int qrow = qbase + wave * 16 + l16;
  v16bf qf[2];
#pragma unroll
  for (int f = 0; f < 2; ++f) {
    const __bf16* qp = Qb + (size_t)qrow * DK + f * 32 + half * 8;
    qf[f] = ld_frag16(qp, qp + 16);
  }

  v8f o[4];
#pragma unroll
  for (int jo = 0; jo < 4; ++jo) o[jo] = vzero8();
  float m_[8], l_[8];
#pragma unroll
  for (int r = 0; r < 8; ++r) { m_[r] = -INFINITY; l_[r] = 0.0f; }

  const int qabs0 = qbase + wave * 16 + half * 8;
  const int nkv = qbase / 64 + 2;

  for (int t = 0; t < nkv; ++t) {
    const int kvb = t * 64;
    __syncthreads();  // previous tile fully consumed
    if (wave == 0)    // DMA K tile [64 kv][64 d] -> LDS with pad stride 72
      tdm_load_2d_bf16(Kb + (size_t)kvb * DK, ks_off, 64, 64,
                       (unsigned)DK, (unsigned)SEQ, (unsigned)DK, 1, 4, 3);
    // All waves stage V transposed while the TDM runs.
    for (int c = tid; c < 512; c += 256) {
      const int row = c >> 3, cc = c & 7;
      v8bf dv = *reinterpret_cast<const v8bf*>(
          Vb + (size_t)(kvb + row) * DMODEL + cc * 8);
#pragma unroll
      for (int e = 0; e < 8; ++e) Vt[(cc * 8 + e) * LDT + row] = dv[e];
    }
    if (wave == 0) __builtin_amdgcn_s_wait_tensorcnt(0);
    __syncthreads();

    // S(16x64) = Q * K_tile^T
    v8f sf[4];
#pragma unroll
    for (int j = 0; j < 4; ++j) {
      const __bf16* p0 = &Ks[(j * 16 + l16) * LDT + half * 8];
      const __bf16* p1 = &Ks[(j * 16 + l16) * LDT + 32 + half * 8];
      v16bf kb0 = ld_frag16(p0, p0 + 16);
      v16bf kb1 = ld_frag16(p1, p1 + 16);
      sf[j] = wmma_bf16(qf[0], kb0, vzero8());
      sf[j] = wmma_bf16(qf[1], kb1, sf[j]);
    }

    // Scale, causal mask, online softmax.
    float p[4][8], rowmax[8];
#pragma unroll
    for (int r = 0; r < 8; ++r) rowmax[r] = -INFINITY;
#pragma unroll
    for (int j = 0; j < 4; ++j) {
      const int kvp = kvb + j * 16 + l16;
#pragma unroll
      for (int r = 0; r < 8; ++r) {
        float v = sf[j][r] * 0.125f;
        if (kvp > qabs0 + r) v = -INFINITY;
        p[j][r] = v;
        rowmax[r] = fmaxf(rowmax[r], v);
      }
    }
#pragma unroll
    for (int off = 1; off < 16; off <<= 1)
#pragma unroll
      for (int r = 0; r < 8; ++r)
        rowmax[r] = fmaxf(rowmax[r], __shfl_xor(rowmax[r], off, 32));

    float corr[8], rs[8];
#pragma unroll
    for (int r = 0; r < 8; ++r) {
      const float mn = fmaxf(m_[r], rowmax[r]);
      corr[r] = __expf(m_[r] - mn);
      m_[r] = mn;
      rs[r] = 0.0f;
    }
#pragma unroll
    for (int j = 0; j < 4; ++j)
#pragma unroll
      for (int r = 0; r < 8; ++r) {
        p[j][r] = __expf(p[j][r] - m_[r]);
        rs[r] += p[j][r];
      }
#pragma unroll
    for (int off = 1; off < 16; off <<= 1)
#pragma unroll
      for (int r = 0; r < 8; ++r) rs[r] += __shfl_xor(rs[r], off, 32);
#pragma unroll
    for (int r = 0; r < 8; ++r) l_[r] = l_[r] * corr[r] + rs[r];
#pragma unroll
    for (int jo = 0; jo < 4; ++jo)
#pragma unroll
      for (int r = 0; r < 8; ++r) o[jo][r] *= corr[r];

    // C-layout P -> LDS -> A-layout fragments.
    __bf16* Pw = &Pl[wave * 16 * LDT];
#pragma unroll
    for (int j = 0; j < 4; ++j)
#pragma unroll
      for (int r = 0; r < 8; ++r)
        Pw[(half * 8 + r) * LDT + j * 16 + l16] = (__bf16)p[j][r];
    __syncthreads();

    v16bf pa[2];
#pragma unroll
    for (int f = 0; f < 2; ++f) {
      const __bf16* pr = &Pw[l16 * LDT + f * 32];
      pa[f] = ld_frag16(pr + half * 8, pr + 16 + half * 8);
    }
#pragma unroll
    for (int jo = 0; jo < 4; ++jo) {
      const __bf16* p0 = &Vt[(jo * 16 + l16) * LDT + half * 8];
      const __bf16* p1 = &Vt[(jo * 16 + l16) * LDT + 32 + half * 8];
      v16bf vb0 = ld_frag16(p0, p0 + 16);
      v16bf vb1 = ld_frag16(p1, p1 + 16);
      o[jo] = wmma_bf16(pa[0], vb0, o[jo]);
      o[jo] = wmma_bf16(pa[1], vb1, o[jo]);
    }
  }

#pragma unroll
  for (int jo = 0; jo < 4; ++jo)
#pragma unroll
    for (int r = 0; r < 8; ++r) {
      const int q = qabs0 + r;
      const int d = jo * 16 + l16;
      O[((size_t)b * SEQ + q) * DMODEL + h * DK + d] =
          (__bf16)(o[jo][r] / l_[r]);
    }
}

// ---------------------------------------------------------------------------
// Host launch
// ---------------------------------------------------------------------------
extern "C" void kernel_launch(void* const* d_in, const int* in_sizes, int n_in,
                              void* d_out, int out_size, void* d_ws, size_t ws_size,
                              hipStream_t stream) {
  const float* x = (const float*)d_in[0];
  const float* Wq = (const float*)d_in[1];
  const float* Wk = (const float*)d_in[2];
  const float* Wv = (const float*)d_in[3];
  const float* Wo = (const float*)d_in[4];
  const int* tpos = (const int*)d_in[5];

  const size_t NX = (size_t)MROWS * DMODEL;
  const size_t NW = (size_t)DMODEL * DMODEL;

  char* ws = (char*)d_ws;
  __bf16* xb = (__bf16*)(ws);
  __bf16* wb = (__bf16*)(ws + 8ull * 1024 * 1024);
  __bf16* qp = (__bf16*)(ws + 16ull * 1024 * 1024);
  __bf16* kp = (__bf16*)(ws + 24ull * 1024 * 1024);
  __bf16* vp = (__bf16*)(ws + 32ull * 1024 * 1024);
  __bf16* qr = (__bf16*)(ws + 40ull * 1024 * 1024);
  __bf16* kr = (__bf16*)(ws + 48ull * 1024 * 1024);
  __bf16* ao = qp;  // reuse projection-Q buffer for attention output

  f32_to_bf16_kernel<<<(int)((NX + 255) / 256), 256, 0, stream>>>(x, xb, (int)NX);
  f32_to_bf16_kernel<<<(int)((NW + 255) / 256), 256, 0, stream>>>(Wq, wb + 0 * NW, (int)NW);
  f32_to_bf16_kernel<<<(int)((NW + 255) / 256), 256, 0, stream>>>(Wk, wb + 1 * NW, (int)NW);
  f32_to_bf16_kernel<<<(int)((NW + 255) / 256), 256, 0, stream>>>(Wv, wb + 2 * NW, (int)NW);
  f32_to_bf16_kernel<<<(int)((NW + 255) / 256), 256, 0, stream>>>(Wo, wb + 3 * NW, (int)NW);

  dim3 gg(DMODEL / 128, MROWS / 128);
  gemm_nt_kernel<false><<<gg, 256, 0, stream>>>(xb, wb + 0 * NW, qp, MROWS, DMODEL, DMODEL);
  gemm_nt_kernel<false><<<gg, 256, 0, stream>>>(xb, wb + 1 * NW, kp, MROWS, DMODEL, DMODEL);
  gemm_nt_kernel<false><<<gg, 256, 0, stream>>>(xb, wb + 2 * NW, vp, MROWS, DMODEL, DMODEL);

  const int nrope = BSZ * NHEAD * SEQ * (DK / 2);
  rope_kernel<<<(nrope + 255) / 256, 256, 0, stream>>>(qp, qr, tpos);
  rope_kernel<<<(nrope + 255) / 256, 256, 0, stream>>>(kp, kr, tpos);

  dim3 ga(SEQ / 128, BSZ * NHEAD);
  flash_attn_kernel<<<ga, 256, 0, stream>>>(qr, kr, vp, ao);

  gemm_nt_kernel<true><<<gg, 256, 0, stream>>>(ao, wb + 3 * NW, d_out, MROWS, DMODEL, DMODEL);
}
